// NerfusionRendererTestSDF_86148454023219
// MI455X (gfx1250) — matrix operand order repaired and checked by reading, not compiled, not run on hardware
//
#include <hip/hip_runtime.h>
#include <hip/hip_bf16.h>
#include <math.h>

#define N_RAYS 16384
#define MAX_HITS 16
#define D_VOX 16
#define HID 64
#define FEAT 15
#define BETA 0.1f
#define MSAMP 64
#define RAYS_PER_BLOCK 8
#define NFRAG 12

typedef __attribute__((ext_vector_type(16))) _Float16 v16h;
typedef __attribute__((ext_vector_type(8)))  _Float16 v8h;
typedef __attribute__((ext_vector_type(8)))  float    v8f;

#if __has_builtin(__builtin_amdgcn_wmma_f16_16x16x32_f16)
#define HAVE_WMMA_F16D 1
#else
#define HAVE_WMMA_F16D 0
#endif

static __device__ __forceinline__ v8f wmma_f32d(v16h a, v16h b, v8f c) {
  return __builtin_amdgcn_wmma_f32_16x16x32_f16(false, a, false, b, (short)0, c,
                                                false, false);
}

// Single-instruction packed ReLU per dword: v_pk_max_num_f16 d, s, 0.
// Inline asm avoids the canonicalize (max(x,x)) clang emits for llvm.maxnum.
// volatile: keeps the relu statements in program order so the sched_barrier
// pattern below provides the required WMMA->VALU hazard distance (ISA 7.12.1).
static __device__ __forceinline__ v8h relu8(v8h x) {
  union { v8h h; unsigned int u[4]; } v;
  v.h = x;
#pragma unroll
  for (int i = 0; i < 4; ++i)
    asm volatile("v_pk_max_num_f16 %0, %1, 0" : "=v"(v.u[i]) : "v"(v.u[i]));
  return v.h;
}

// Merge two contiguous 8-half LDS chunks into one 16-half A fragment.
// 16-bit A-matrix 16x32 layout (ISA 7.12.2): lane half `hi` owns K runs
// [hi*8, hi*8+7] and [16+hi*8, 16+hi*8+7] -> two aligned b128 DS loads.
static __device__ __forceinline__ v16h load_afrag_lds(const _Float16* rowp, int hi) {
  const v8h lo = *reinterpret_cast<const v8h*>(rowp + hi * 8);
  const v8h up = *reinterpret_cast<const v8h*>(rowp + 16 + hi * 8);
  return __builtin_shufflevector(lo, up, 0, 1, 2, 3, 4, 5, 6, 7,
                                 8, 9, 10, 11, 12, 13, 14, 15);
}

static __device__ __forceinline__ float sigmoidf(float x) {
  return 1.0f / (1.0f + __expf(-x));
}

__global__ __launch_bounds__(256)
void nerfusion_render_kernel(const float* __restrict__ rays_o,
                             const float* __restrict__ rays_d,
                             const int*   __restrict__ vox_idx,
                             const float* __restrict__ t_near,
                             const float* __restrict__ t_far,
                             const float* __restrict__ embed,
                             const float* __restrict__ W1, const float* __restrict__ b1,
                             const float* __restrict__ W2, const float* __restrict__ b2,
                             const float* __restrict__ Wr1, const float* __restrict__ br1,
                             const float* __restrict__ Wr2, const float* __restrict__ br2,
                             float* __restrict__ out) {
  // B fragment images (shared by all 8 waves): [frag][lane][elem], f16
  __shared__ __align__(16) _Float16 frag_lds[NFRAG * 32 * 16];
  __shared__ __align__(16) _Float16 h_lds [RAYS_PER_BLOCK][16 * HID]; // hidden (f16)
  __shared__ __align__(16) _Float16 xr_lds[RAYS_PER_BLOCK][16 * 32];  // [dir,feats,1] K=32
  __shared__ float sdf_s[RAYS_PER_BLOCK][MSAMP];
  __shared__ float t_s  [RAYS_PER_BLOCK][MSAMP];
  __shared__ float d_s  [RAYS_PER_BLOCK][MSAMP];
  __shared__ float m_s  [RAYS_PER_BLOCK][MSAMP];
  __shared__ float rgb_s[RAYS_PER_BLOCK][MSAMP * 3];

  const int lane = threadIdx.x & 31;
  const int wv   = threadIdx.x >> 5;   // wave id in block = ray in block
  const int ray  = blockIdx.x * RAYS_PER_BLOCK + wv;
  const int n    = lane & 15;          // B/C column, A row
  const int hi   = lane >> 4;

  // ---- cooperative B-fragment staging (once per block) --------------------
  // Frag ids: 0-3 W1(+b1 at slot 3), 4-5 W2 (K slices), 6-9 Wr1(+br1 at slot 18),
  // 10-11 Wr2 (K slices, cols padded 3->16).
  // B layout: lane = column N (lane&15), K-slot = (e&7) + 8*(lane>>4) + 16*(e>>3).
  // Layer-1 x permutation: slot 0..2 = pts, slot 3 = 1.0 (bias), slot 8..23 =
  // embed[0..15] (row = slot-5), rest zero.
  for (int i = threadIdx.x; i < NFRAG * 32 * 16; i += 256) {
    const int e    = i & 15;
    const int ln   = (i >> 4) & 31;
    const int f    = i >> 9;
    const int n2   = ln & 15;
    const int slot = (e & 7) + ((ln >> 4) << 3) + ((e >> 3) << 4);
    float val = 0.0f;
    if (f < 4) {
      const int col = f * 16 + n2;
      if (slot < 3)                      val = W1[slot * HID + col];
      else if (slot == 3)                val = b1[col];
      else if (slot >= 8 && slot < 24)   val = W1[(slot - 5) * HID + col];
    } else if (f < 6) {
      val = W2[((f - 4) * 32 + slot) * 16 + n2];
    } else if (f < 10) {
      const int col = (f - 6) * 16 + n2;
      if (slot < 3 + FEAT)               val = Wr1[slot * HID + col];
      else if (slot == 3 + FEAT)         val = br1[col];
    } else {
      if (n2 < 3)                        val = Wr2[((f - 10) * 32 + slot) * 3 + n2];
    }
    frag_lds[i] = (_Float16)val;
  }
  __syncthreads();

  const _Float16* fb = frag_lds;
  v16h bW1[4], bW2[2], bWr1[4], bWr2[2];
#pragma unroll
  for (int nt = 0; nt < 4; ++nt) bW1[nt]  = *reinterpret_cast<const v16h*>(fb + (nt    ) * 512 + lane * 16);
#pragma unroll
  for (int ks = 0; ks < 2; ++ks) bW2[ks]  = *reinterpret_cast<const v16h*>(fb + (4 + ks) * 512 + lane * 16);
#pragma unroll
  for (int nt = 0; nt < 4; ++nt) bWr1[nt] = *reinterpret_cast<const v16h*>(fb + (6 + nt) * 512 + lane * 16);
#pragma unroll
  for (int ks = 0; ks < 2; ++ks) bWr2[ks] = *reinterpret_cast<const v16h*>(fb + (10 + ks) * 512 + lane * 16);

  const float b2v  = b2[n];
  const float br2v = (n < 3) ? br2[n] : 0.0f;

  const float ox = rays_o[ray * 3 + 0], oy = rays_o[ray * 3 + 1], oz = rays_o[ray * 3 + 2];
  const float dx = rays_d[ray * 3 + 0], dy = rays_d[ray * 3 + 1], dz = rays_d[ray * 3 + 2];

  // prefetch this ray's embed rows (global_prefetch_b8)
  {
    const int vox0 = vox_idx[ray * MAX_HITS + n];
    const long er  = (long)(vox0 < 0 ? 0 : vox0) * D_VOX;
    __builtin_prefetch(&embed[er], 0, 0);
  }

  // constant xr columns (dir, bias-1 at slot 18, zero pad), once per ray
  if (hi == 0) {
    _Float16* row = &xr_lds[wv][n * 32];
    row[0] = (_Float16)dx;
    row[1] = (_Float16)dy;
    row[2] = (_Float16)dz;
    row[3 + FEAT] = (_Float16)1.0f;     // bias lane for layer 3
#pragma unroll
    for (int k = 4 + FEAT; k < 32; ++k) row[k] = (_Float16)0.0f;
  }

  for (int tile = 0; tile < 4; ++tile) {
    const int   s    = tile * 16 + n;   // sample (row) for this lane
    const int   hit  = s >> 2;
    const int   sub  = s & 3;
    const float tn   = t_near[ray * MAX_HITS + hit];
    const float tf   = t_far [ray * MAX_HITS + hit];
    const float seg  = tf - tn;
    const float tv   = tn + seg * (((float)sub + 0.5f) * 0.25f);
    const float dist = seg * 0.25f;
    const int   vox  = vox_idx[ray * MAX_HITS + hit];
    const float mk   = (vox >= 0) ? 1.0f : 0.0f;
    const long  erow = (long)(vox < 0 ? 0 : vox) * D_VOX;

    const float px = ox + tv * dx, py = oy + tv * dy, pz = oz + tv * dz;
    if (hi == 0) { t_s[wv][s] = tv; d_s[wv][s] = dist; m_s[wv][s] = mk; }

    // ---- layer 1 A fragment (permuted K layout, branchless) ----
    // hi==1 lanes own embed[0..7], hi==0 lanes own embed[8..15]; 2 x float4.
    // hi==0 slot 3 carries the constant 1.0 bias lane.
    const float4* erowp = reinterpret_cast<const float4*>(embed + erow);
    const float4 q0 = erowp[hi ? 0 : 2];
    const float4 q1 = erowp[hi ? 1 : 3];
    v16h a1;
    a1[0]  = (_Float16)(hi ? q0.x : px);
    a1[1]  = (_Float16)(hi ? q0.y : py);
    a1[2]  = (_Float16)(hi ? q0.z : pz);
    a1[3]  = (_Float16)(hi ? q0.w : 1.0f);
    a1[4]  = (_Float16)(hi ? q1.x : 0.0f);
    a1[5]  = (_Float16)(hi ? q1.y : 0.0f);
    a1[6]  = (_Float16)(hi ? q1.z : 0.0f);
    a1[7]  = (_Float16)(hi ? q1.w : 0.0f);
    a1[8]  = (_Float16)(hi ? 0.0f : q0.x);
    a1[9]  = (_Float16)(hi ? 0.0f : q0.y);
    a1[10] = (_Float16)(hi ? 0.0f : q0.z);
    a1[11] = (_Float16)(hi ? 0.0f : q0.w);
    a1[12] = (_Float16)(hi ? 0.0f : q1.x);
    a1[13] = (_Float16)(hi ? 0.0f : q1.y);
    a1[14] = (_Float16)(hi ? 0.0f : q1.z);
    a1[15] = (_Float16)(hi ? 0.0f : q1.w);

#if HAVE_WMMA_F16D
    // f16 accumulate: D already f16; wmma pass, sched barrier, then relu pass
    // in order => every WMMA's D consumer sits >= 4 issue slots downstream.
    {
      v8h d1[4];
#pragma unroll
      for (int nt = 0; nt < 4; ++nt) {
        v8h z = {};
        d1[nt] = __builtin_amdgcn_wmma_f16_16x16x32_f16(false, a1, false, bW1[nt],
                                                        (short)0, z, false, false);
      }
      __builtin_amdgcn_sched_barrier(0);
#pragma unroll
      for (int nt = 0; nt < 4; ++nt) d1[nt] = relu8(d1[nt]);
#pragma unroll
      for (int nt = 0; nt < 4; ++nt)
#pragma unroll
        for (int e = 0; e < 8; ++e)
          h_lds[wv][(hi * 8 + e) * HID + nt * 16 + n] = d1[nt][e];
    }
#else
    v8f c1t[4] = {v8f{}, v8f{}, v8f{}, v8f{}};
#pragma unroll
    for (int nt = 0; nt < 4; ++nt) c1t[nt] = wmma_f32d(a1, bW1[nt], c1t[nt]);
#pragma unroll
    for (int nt = 0; nt < 4; ++nt)
#pragma unroll
      for (int r = 0; r < 8; ++r) {
        float h = c1t[nt][r];                  // bias already folded in GEMM
        h = h > 0.0f ? h : 0.0f;
        h_lds[wv][(r + hi * 8) * HID + nt * 16 + n] = (_Float16)h;
      }
#endif
    asm volatile("s_wait_dscnt 0x0" ::: "memory");

    // ---- layer 2: so = h @ W2 + b2 (K=64 as 2 slices, vector DS reads) ----
    v8f cso = {};
#pragma unroll
    for (int ks = 0; ks < 2; ++ks) {
      const v16h a2 = load_afrag_lds(&h_lds[wv][n * HID + ks * 32], hi);
      cso = wmma_f32d(a2, bW2[ks], cso);
    }

    // feats -> xr_lds, sdf col 0 -> sdf_s
#pragma unroll
    for (int r = 0; r < 8; ++r) {
      const int row = r + hi * 8;
      const float so = cso[r] + b2v;
      if (n == 0) sdf_s[wv][tile * 16 + row] = so;          // lanes 0,16 cover all rows
      else        xr_lds[wv][row * 32 + (n + 2)] = (_Float16)so;
    }
    asm volatile("s_wait_dscnt 0x0" ::: "memory");

    // ---- layer 3: hr = relu(xr @ Wr1 + br1) (bias folded at slot 18) ----
    const v16h a3 = load_afrag_lds(&xr_lds[wv][n * 32], hi);
#if HAVE_WMMA_F16D
    {
      v8h d3[4];
#pragma unroll
      for (int nt = 0; nt < 4; ++nt) {
        v8h z = {};
        d3[nt] = __builtin_amdgcn_wmma_f16_16x16x32_f16(false, a3, false, bWr1[nt],
                                                        (short)0, z, false, false);
      }
      __builtin_amdgcn_sched_barrier(0);
#pragma unroll
      for (int nt = 0; nt < 4; ++nt) d3[nt] = relu8(d3[nt]);
#pragma unroll
      for (int nt = 0; nt < 4; ++nt)
#pragma unroll
        for (int e = 0; e < 8; ++e)
          h_lds[wv][(hi * 8 + e) * HID + nt * 16 + n] = d3[nt][e];
    }
#else
    v8f c3t[4] = {v8f{}, v8f{}, v8f{}, v8f{}};
#pragma unroll
    for (int nt = 0; nt < 4; ++nt) c3t[nt] = wmma_f32d(a3, bWr1[nt], c3t[nt]);
#pragma unroll
    for (int nt = 0; nt < 4; ++nt)
#pragma unroll
      for (int r = 0; r < 8; ++r) {
        float h = c3t[nt][r];
        h = h > 0.0f ? h : 0.0f;
        h_lds[wv][(r + hi * 8) * HID + nt * 16 + n] = (_Float16)h;
      }
#endif
    asm volatile("s_wait_dscnt 0x0" ::: "memory");

    // ---- layer 4: rgb = sigmoid(hr @ Wr2 + br2) (N padded 3->16) ----
    v8f crgb = {};
#pragma unroll
    for (int ks = 0; ks < 2; ++ks) {
      const v16h a4 = load_afrag_lds(&h_lds[wv][n * HID + ks * 32], hi);
      crgb = wmma_f32d(a4, bWr2[ks], crgb);
    }
    if (n < 3) {
#pragma unroll
      for (int r = 0; r < 8; ++r) {
        const int row = r + hi * 8;
        rgb_s[wv][(tile * 16 + row) * 3 + n] = sigmoidf(crgb[r] + br2v);
      }
    }
    asm volatile("s_wait_dscnt 0x0" ::: "memory");
  }

  // ---- volumetric composite: 2 samples per lane, shuffle scan + reductions ----
  const int s0 = 2 * lane, s1 = 2 * lane + 1;
  const float inv_beta = 1.0f / BETA;
  const float sd0 = sdf_s[wv][s0], sd1 = sdf_s[wv][s1];
  const float m0  = m_s[wv][s0],   m1  = m_s[wv][s1];
  const float di0 = d_s[wv][s0],   di1 = d_s[wv][s1];
  const float tv0 = t_s[wv][s0],   tv1 = t_s[wv][s1];

  const float fe0 = inv_beta * sigmoidf(-sd0 * inv_beta) * di0 * m0;
  const float fe1 = inv_beta * sigmoidf(-sd1 * inv_beta) * di1 * m1;

  const float pair = fe0 + fe1;
  float scan = pair;                       // inclusive scan of pair sums over lanes
#pragma unroll
  for (int d = 1; d < 32; d <<= 1) {
    const float v = __shfl_up(scan, d, 32);
    if (lane >= d) scan += v;
  }
  const float excl = scan - pair;          // exclusive prefix of free_energy at s0
  const float T0 = __expf(-excl);
  const float T1 = __expf(-(excl + fe0));
  const float w0 = (1.0f - __expf(-fe0)) * T0;
  const float w1 = (1.0f - __expf(-fe1)) * T1;

  const float r0 = rgb_s[wv][s0 * 3 + 0], g0 = rgb_s[wv][s0 * 3 + 1], u0 = rgb_s[wv][s0 * 3 + 2];
  const float r1 = rgb_s[wv][s1 * 3 + 0], g1 = rgb_s[wv][s1 * 3 + 1], u1 = rgb_s[wv][s1 * 3 + 2];

  float accP = w0 + w1;
  float depP = w0 * tv0 + w1 * tv1;
  float rP = w0 * r0 + w1 * r1;
  float gP = w0 * g0 + w1 * g1;
  float uP = w0 * u0 + w1 * u1;
  float mP = m0 + m1;
#pragma unroll
  for (int d = 16; d >= 1; d >>= 1) {
    accP += __shfl_xor(accP, d, 32);
    depP += __shfl_xor(depP, d, 32);
    rP   += __shfl_xor(rP, d, 32);
    gP   += __shfl_xor(gP, d, 32);
    uP   += __shfl_xor(uP, d, 32);
    mP   += __shfl_xor(mP, d, 32);
  }
  const bool hm = mP > 0.0f;               // recomputed ray_hits

  const long N = N_RAYS;
  float* w_out   = out + 6  * N;           // weights [N,64]
  float* sdf_out = out + 70 * N;           // sdf     [N,64]
  w_out  [(long)ray * 64 + s0] = hm ? w0 : 0.0f;
  w_out  [(long)ray * 64 + s1] = hm ? w1 : 0.0f;
  sdf_out[(long)ray * 64 + s0] = hm ? sd0 * m0 : 1.0f;
  sdf_out[(long)ray * 64 + s1] = hm ? sd1 * m1 : 1.0f;

  if (lane == 0) {
    float rr = rP, gg = gP, uu = uP, depth = depP, acc = accP;
    const float eps = 1e-10f;
    float disp = 1.0f / fmaxf(eps, depth / fmaxf(acc, eps));
    if (!hm) { rr = gg = uu = 0.0f; depth = 0.0f; acc = 0.0f; disp = 1000.0f; }
    out[ray * 3 + 0] = rr;
    out[ray * 3 + 1] = gg;
    out[ray * 3 + 2] = uu;
    out[3 * N + ray] = depth;
    out[4 * N + ray] = disp;
    out[5 * N + ray] = acc;
  }
}

extern "C" void kernel_launch(void* const* d_in, const int* in_sizes, int n_in,
                              void* d_out, int out_size, void* d_ws, size_t ws_size,
                              hipStream_t stream) {
  const float* rays_o = (const float*)d_in[0];
  const float* rays_d = (const float*)d_in[1];
  const int*   vox    = (const int*)  d_in[2];
  const float* t_near = (const float*)d_in[3];
  const float* t_far  = (const float*)d_in[4];
  // d_in[5] = ray_hits (bool) -- recomputed in-kernel from vox_idx
  const float* embed  = (const float*)d_in[6];
  const float* W1  = (const float*)d_in[7];
  const float* b1  = (const float*)d_in[8];
  const float* W2  = (const float*)d_in[9];
  const float* b2  = (const float*)d_in[10];
  const float* Wr1 = (const float*)d_in[11];
  const float* br1 = (const float*)d_in[12];
  const float* Wr2 = (const float*)d_in[13];
  const float* br2 = (const float*)d_in[14];

  dim3 grid(N_RAYS / RAYS_PER_BLOCK);
  dim3 block(256);
  hipLaunchKernelGGL(nerfusion_render_kernel, grid, block, 0, stream,
                     rays_o, rays_d, vox, t_near, t_far, embed,
                     W1, b1, W2, b2, Wr1, br1, Wr2, br2, (float*)d_out);
}
